// ModuleNet_70111046140090
// MI455X (gfx1250) — matrix operand order repaired and checked by reading, not compiled, not run on hardware
//
#include <hip/hip_runtime.h>
#include <hip/hip_bf16.h>
#include <stdint.h>

typedef __attribute__((ext_vector_type(16))) _Float16 v16h;
typedef __attribute__((ext_vector_type(8)))  float    v8f;
typedef __attribute__((ext_vector_type(4)))  float    v4f;

// A-fragment conversion helper: 16 f32 (two 8-float runs) -> v16h
__device__ __forceinline__ v16h make_afrag(const float* rowp) {
    v4f l0 = *(const v4f*)(rowp);
    v4f l1 = *(const v4f*)(rowp + 4);
    v4f h0 = *(const v4f*)(rowp + 16);
    v4f h1 = *(const v4f*)(rowp + 20);
    v16h a;
    a[0]  = (_Float16)l0[0]; a[1]  = (_Float16)l0[1];
    a[2]  = (_Float16)l0[2]; a[3]  = (_Float16)l0[3];
    a[4]  = (_Float16)l1[0]; a[5]  = (_Float16)l1[1];
    a[6]  = (_Float16)l1[2]; a[7]  = (_Float16)l1[3];
    a[8]  = (_Float16)h0[0]; a[9]  = (_Float16)h0[1];
    a[10] = (_Float16)h0[2]; a[11] = (_Float16)h0[3];
    a[12] = (_Float16)h1[0]; a[13] = (_Float16)h1[1];
    a[14] = (_Float16)h1[2]; a[15] = (_Float16)h1[3];
    return a;
}

// ---------------------------------------------------------------------------
// W pre-pass: convert W[512][128][128] f32 -> f16, stored in exact WMMA
// A-fragment order so Phase A loads one contiguous 32B fragment per lane.
// Fragment address: Wh + ((((mid*8 + t)*4 + kk)*32) + lane)*16 halves.
// ---------------------------------------------------------------------------
__global__ __launch_bounds__(128) void convert_W_kernel(
    const float* __restrict__ Wn, _Float16* __restrict__ Wh)
{
    const int bid  = blockIdx.x;
    const int mid  = bid >> 3;
    const int t    = bid & 7;
    const int kk   = threadIdx.x >> 5;
    const int lane = threadIdx.x & 31;
    const int col  = lane & 15;
    const int hf   = lane >> 4;

    const float* rowp = Wn + ((size_t)mid * 128 + t * 16 + col) * 128 + kk * 32 + hf * 8;
    v16h a = make_afrag(rowp);
    _Float16* dst = Wh + ((((size_t)mid * 8 + t) * 4 + kk) * 32 + lane) * 16;
    *(v16h*)dst = a;
}

// ---------------------------------------------------------------------------
// Phase A (fast): per-(b,p) chain of 4 gathered 128x128 layers on (x,y).
// One wave32 per pair; 8 waves per block; 512 blocks = 4096 pairs.
// A-fragments double-buffered across M-tiles: tile t+1's 8 b128 loads are
// issued before tile t's 4-WMMA chain so waits become partial, keeping the
// L2 load stream saturated instead of load->wait_all->wmma serialization.
// ---------------------------------------------------------------------------
__global__ __launch_bounds__(256) void path_chain_fast_kernel(
    const int*      __restrict__ start_ids,
    const int*      __restrict__ end_ids,
    const int*      __restrict__ module_ids,
    const float*    __restrict__ embed,
    const _Float16* __restrict__ Wh,   // fragment-swizzled f16 W
    const float*    __restrict__ bn,
    float*          __restrict__ feat) // [4096][256]
{
    __shared__ float vec[8][2][2][128];   // [wave][buf][x/y][E] = 16 KB

    const int tid  = threadIdx.x;
    const int w    = tid >> 5;
    const int lane = tid & 31;
    const int col  = lane & 15;
    const int hf   = lane >> 4;
    const int pair = blockIdx.x * 8 + w;

    {
        const int sid = start_ids[pair];
        const int eid = end_ids[pair];
        v4f vx = *(const v4f*)(embed + (size_t)sid * 128 + lane * 4);
        v4f vy = *(const v4f*)(embed + (size_t)eid * 128 + lane * 4);
        *(v4f*)(&vec[w][0][0][lane * 4]) = vx;
        *(v4f*)(&vec[w][0][1][lane * 4]) = vy;
    }
    int cur = 0;
    asm volatile("" ::: "memory");

    for (int l = 0; l < 4; ++l) {
        const int mid = module_ids[pair * 4 + l];
        const _Float16* Wm = Wh + (size_t)mid * (128 * 128);
        const float*    bm = bn + (size_t)mid * 128;

        // B fragments (x in col 0, y in col 1) for the 4 K-steps
        v16h bfrag[4];
        #pragma unroll
        for (int kk = 0; kk < 4; ++kk) {
            v16h f;
            if (col < 2) {
                const float* src = &vec[w][cur][col][kk * 32 + hf * 8];
                #pragma unroll
                for (int j = 0; j < 16; ++j) f[j] = (_Float16)src[j + (j & 8)];
            } else {
                #pragma unroll
                for (int j = 0; j < 16; ++j) f[j] = (_Float16)0.0f;
            }
            bfrag[kk] = f;
        }
        asm volatile("" ::: "memory");

        // ---- software-pipelined M-tile loop (double-buffered A frags) ----
        v16h afr[2][4];
        v4f  bia[2][2];
        #pragma unroll
        for (int kk = 0; kk < 4; ++kk)
            afr[0][kk] = *(const v16h*)(Wm + ((size_t)kk * 32 + lane) * 16);
        bia[0][0] = *(const v4f*)(bm + hf * 8);
        bia[0][1] = *(const v4f*)(bm + hf * 8 + 4);

        #pragma unroll
        for (int t = 0; t < 8; ++t) {
            const int cb = t & 1, nb = cb ^ 1;
            if (t < 7) {
                #pragma unroll
                for (int kk = 0; kk < 4; ++kk)
                    afr[nb][kk] = *(const v16h*)(Wm + (((size_t)(t + 1) * 4 + kk) * 32 + lane) * 16);
                bia[nb][0] = *(const v4f*)(bm + (t + 1) * 16 + hf * 8);
                bia[nb][1] = *(const v4f*)(bm + (t + 1) * 16 + hf * 8 + 4);
            }
            v8f acc;
            acc[0] = bia[cb][0][0]; acc[1] = bia[cb][0][1];
            acc[2] = bia[cb][0][2]; acc[3] = bia[cb][0][3];
            acc[4] = bia[cb][1][0]; acc[5] = bia[cb][1][1];
            acc[6] = bia[cb][1][2]; acc[7] = bia[cb][1][3];
            #pragma unroll
            for (int kk = 0; kk < 4; ++kk)
                acc = __builtin_amdgcn_wmma_f32_16x16x32_f16(
                          false, afr[cb][kk], false, bfrag[kk], (short)0, acc, false, false);
            if (col < 2) {
                float* dst = &vec[w][cur ^ 1][col][t * 16 + hf * 8];
                #pragma unroll
                for (int r = 0; r < 8; ++r) dst[r] = acc[r];
            }
        }
        cur ^= 1;
        asm volatile("" ::: "memory");
    }

    {
        const float* src = &vec[w][cur][0][0];
        float* dst = feat + (size_t)pair * 256;
        #pragma unroll
        for (int i = 0; i < 8; ++i) dst[lane * 8 + i] = src[lane * 8 + i];
    }
}

// ---------------------------------------------------------------------------
// Phase A (fallback, ws too small): direct f32 W with on-the-fly conversion.
// ---------------------------------------------------------------------------
__global__ __launch_bounds__(256) void path_chain_kernel(
    const int*   __restrict__ start_ids,
    const int*   __restrict__ end_ids,
    const int*   __restrict__ module_ids,
    const float* __restrict__ embed,
    const float* __restrict__ Wn,
    const float* __restrict__ bn,
    float*       __restrict__ feat)
{
    __shared__ float vec[8][2][2][128];

    const int tid  = threadIdx.x;
    const int w    = tid >> 5;
    const int lane = tid & 31;
    const int col  = lane & 15;
    const int hf   = lane >> 4;
    const int pair = blockIdx.x * 8 + w;

    {
        const int sid = start_ids[pair];
        const int eid = end_ids[pair];
        v4f vx = *(const v4f*)(embed + (size_t)sid * 128 + lane * 4);
        v4f vy = *(const v4f*)(embed + (size_t)eid * 128 + lane * 4);
        *(v4f*)(&vec[w][0][0][lane * 4]) = vx;
        *(v4f*)(&vec[w][0][1][lane * 4]) = vy;
    }
    int cur = 0;
    asm volatile("" ::: "memory");

    for (int l = 0; l < 4; ++l) {
        const int mid = module_ids[pair * 4 + l];
        const float* Wm = Wn + (size_t)mid * (128 * 128);
        const float* bm = bn + (size_t)mid * 128;

        v16h bfrag[4];
        #pragma unroll
        for (int kk = 0; kk < 4; ++kk) {
            v16h f;
            if (col < 2) {
                const float* src = &vec[w][cur][col][kk * 32 + hf * 8];
                #pragma unroll
                for (int j = 0; j < 16; ++j) f[j] = (_Float16)src[j + (j & 8)];
            } else {
                #pragma unroll
                for (int j = 0; j < 16; ++j) f[j] = (_Float16)0.0f;
            }
            bfrag[kk] = f;
        }
        asm volatile("" ::: "memory");

        #pragma unroll 2
        for (int t = 0; t < 8; ++t) {
            v8f acc;
            {
                const float* bp = bm + t * 16 + hf * 8;
                v4f b0 = *(const v4f*)(bp);
                v4f b1 = *(const v4f*)(bp + 4);
                acc[0] = b0[0]; acc[1] = b0[1]; acc[2] = b0[2]; acc[3] = b0[3];
                acc[4] = b1[0]; acc[5] = b1[1]; acc[6] = b1[2]; acc[7] = b1[3];
            }
            #pragma unroll
            for (int kk = 0; kk < 4; ++kk) {
                const float* rowp = Wm + (size_t)(t * 16 + col) * 128 + kk * 32 + hf * 8;
                v16h a = make_afrag(rowp);
                acc = __builtin_amdgcn_wmma_f32_16x16x32_f16(
                          false, a, false, bfrag[kk], (short)0, acc, false, false);
            }
            if (col < 2) {
                float* dst = &vec[w][cur ^ 1][col][t * 16 + hf * 8];
                #pragma unroll
                for (int r = 0; r < 8; ++r) dst[r] = acc[r];
            }
        }
        cur ^= 1;
        asm volatile("" ::: "memory");
    }

    {
        const float* src = &vec[w][cur][0][0];
        float* dst = feat + (size_t)pair * 256;
        #pragma unroll
        for (int i = 0; i < 8; ++i) dst[lane * 8 + i] = src[lane * 8 + i];
    }
}

// ---------------------------------------------------------------------------
// Phase B1: counts-weighted pooling -> pooledT [feature=256][batch=256]
// ---------------------------------------------------------------------------
__global__ __launch_bounds__(256) void pool_kernel(
    const float* __restrict__ feat,
    const int*   __restrict__ counts,
    float*       __restrict__ pooledT)
{
    const int b = blockIdx.x;
    const int f = threadIdx.x;
    float s = 0.0f, cs = 0.0f;
    #pragma unroll
    for (int p = 0; p < 16; ++p) {
        float c = (float)counts[b * 16 + p];
        s  += c * feat[(size_t)(b * 16 + p) * 256 + f];
        cs += c;
    }
    pooledT[(size_t)f * 256 + b] = s / cs;
}

// ---------------------------------------------------------------------------
// Phase B2/B3: OutT[M][256] = act( A[M][K] * BT[K][256] + bias[M] )
// ---------------------------------------------------------------------------
__global__ __launch_bounds__(256) void gemm_bias_act_kernel(
    const float* __restrict__ A,     // [M][K]
    const float* __restrict__ BT,    // [K][256]
    const float* __restrict__ bias,  // [M]
    float*       __restrict__ OutT,  // [M][256]
    int M, int K, int do_relu)
{
    __shared__ uint32_t bst[256 * 16];   // packed f16 K-pairs, 16 KB

    const int tid  = threadIdx.x;
    const int w    = tid >> 5;
    const int lane = tid & 31;
    const int col  = lane & 15;
    const int hf   = lane >> 4;
    const int nbase = blockIdx.y * 16;
    const int mbase = blockIdx.x * 128 + w * 16;

    for (int k2 = tid; k2 < (K >> 1); k2 += 256) {
        const float* r0 = BT + (size_t)(2 * k2) * 256 + nbase;
        const float* r1 = r0 + 256;
        #pragma unroll
        for (int n = 0; n < 16; ++n) {
            union { _Float16 h[2]; uint32_t u; } pk;
            pk.h[0] = (_Float16)r0[n];
            pk.h[1] = (_Float16)r1[n];
            bst[k2 * 16 + n] = pk.u;
        }
    }
    __syncthreads();

    v8f acc;
    {
        const float* bp = bias + mbase + hf * 8;
        v4f b0 = *(const v4f*)(bp);
        v4f b1 = *(const v4f*)(bp + 4);
        acc[0] = b0[0]; acc[1] = b0[1]; acc[2] = b0[2]; acc[3] = b0[3];
        acc[4] = b1[0]; acc[5] = b1[1]; acc[6] = b1[2]; acc[7] = b1[3];
    }

    const int ksteps = K >> 5;
    for (int kk = 0; kk < ksteps; ++kk) {
        const float* rowp = A + (size_t)(mbase + col) * K + kk * 32 + hf * 8;
        v16h a = make_afrag(rowp);
        v16h bf;
        #pragma unroll
        for (int j = 0; j < 8; ++j) {
            union { uint32_t u; _Float16 h[2]; } pk;
            pk.u = bst[(kk * 16 + hf * 4 + j + (j & 4)) * 16 + col];
            bf[2 * j]     = pk.h[0];
            bf[2 * j + 1] = pk.h[1];
        }
        acc = __builtin_amdgcn_wmma_f32_16x16x32_f16(
                  false, a, false, bf, (short)0, acc, false, false);
    }

    if (do_relu) {
        #pragma unroll
        for (int r = 0; r < 8; ++r) acc[r] = fmaxf(acc[r], 0.0f);
    }
    #pragma unroll
    for (int r = 0; r < 8; ++r)
        OutT[(size_t)(mbase + hf * 8 + r) * 256 + nbase + col] = acc[r];
}

// ---------------------------------------------------------------------------
// Phase B4: out[b] = sigmoid( W3 . H2T[:,b] + b3 )
// ---------------------------------------------------------------------------
__global__ __launch_bounds__(256) void head_kernel(
    const float* __restrict__ H2T,
    const float* __restrict__ W3,
    const float* __restrict__ b3,
    float*       __restrict__ out)
{
    const int b = threadIdx.x;
    float acc = b3[0];
    #pragma unroll 4
    for (int k = 0; k < 256; ++k)
        acc += W3[k] * H2T[(size_t)k * 256 + b];
    out[b] = 1.0f / (1.0f + __expf(-acc));
}

// ---------------------------------------------------------------------------
extern "C" void kernel_launch(void* const* d_in, const int* in_sizes, int n_in,
                              void* d_out, int out_size, void* d_ws, size_t ws_size,
                              hipStream_t stream) {
    const int*   start_ids  = (const int*)  d_in[0];
    const int*   end_ids    = (const int*)  d_in[1];
    const int*   module_ids = (const int*)  d_in[2];
    const int*   counts     = (const int*)  d_in[3];
    const float* embed      = (const float*)d_in[4];
    const float* Wn         = (const float*)d_in[5];
    const float* bn         = (const float*)d_in[6];
    const float* W1         = (const float*)d_in[7];
    const float* b1         = (const float*)d_in[8];
    const float* W2         = (const float*)d_in[9];
    const float* b2         = (const float*)d_in[10];
    const float* W3         = (const float*)d_in[11];
    const float* b3         = (const float*)d_in[12];
    float* out = (float*)d_out;

    char* ws = (char*)d_ws;
    const size_t sz_feat   = (size_t)4096 * 256 * 4;   // 4 MB
    const size_t sz_pooled = (size_t)256 * 256 * 4;    // 256 KB
    const size_t sz_h1     = (size_t)512 * 256 * 4;    // 512 KB
    const size_t sz_h2     = (size_t)256 * 256 * 4;    // 256 KB
    float* feat    = (float*)(ws);
    float* pooledT = (float*)(ws + sz_feat);
    float* H1T     = (float*)(ws + sz_feat + sz_pooled);
    float* H2T     = (float*)(ws + sz_feat + sz_pooled + sz_h1);
    const size_t off_Wh = sz_feat + sz_pooled + sz_h1 + sz_h2;   // 5 MB, 32B aligned
    const size_t sz_Wh  = (size_t)512 * 128 * 128 * 2;           // 16 MB f16

    if (ws_size >= off_Wh + sz_Wh) {
        _Float16* Wh = (_Float16*)(ws + off_Wh);
        convert_W_kernel<<<dim3(512 * 8), dim3(128), 0, stream>>>(Wn, Wh);
        path_chain_fast_kernel<<<dim3(512), dim3(256), 0, stream>>>(
            start_ids, end_ids, module_ids, embed, Wh, bn, feat);
    } else {
        path_chain_kernel<<<dim3(512), dim3(256), 0, stream>>>(
            start_ids, end_ids, module_ids, embed, Wn, bn, feat);
    }

    pool_kernel<<<dim3(256), dim3(256), 0, stream>>>(feat, counts, pooledT);

    gemm_bias_act_kernel<<<dim3(4, 16), dim3(256), 0, stream>>>(
        W1, pooledT, b1, H1T, 512, 256, 1);

    gemm_bias_act_kernel<<<dim3(2, 16), dim3(256), 0, stream>>>(
        W2, H1T, b2, H2T, 256, 512, 1);

    head_kernel<<<dim3(1), dim3(256), 0, stream>>>(H2T, W3, b3, out);
}
